// Batch_KMeans_10668698763637
// MI455X (gfx1250) — matrix operand order repaired and checked by reading, not compiled, not run on hardware
//
#include <hip/hip_runtime.h>

#define N_PTS 32768
#define K_CL  2048
#define D     512
#define DECAYF 0.99f

typedef __attribute__((ext_vector_type(16))) __bf16 v16bf;
typedef __attribute__((ext_vector_type(8)))  float  v8f;

union Frag { uint4 q[2]; v16bf v; };

__device__ __forceinline__ unsigned short f2bf(float f) {
  unsigned int u = __float_as_uint(f);
  u += 0x7FFFu + ((u >> 16) & 1u);        // round-to-nearest-even
  return (unsigned short)(u >> 16);
}
__device__ __forceinline__ float bf2f(unsigned short h) {
  return __uint_as_float(((unsigned int)h) << 16);
}

// ---------------------------------------------------------------- row norms
__global__ __launch_bounds__(256) void k_norms(const float* __restrict__ X,
                                               float* __restrict__ invn) {
  const int w = threadIdx.x >> 5, lane = threadIdx.x & 31;
  const int row = blockIdx.x * 8 + w;
  const float* x = X + row * D;
  float s = 0.f;
#pragma unroll
  for (int j = 0; j < D / 32; ++j) { float v = x[lane + j * 32]; s += v * v; }
#pragma unroll
  for (int m = 16; m; m >>= 1) s += __shfl_xor(s, m, 32);
  if (lane == 0) invn[row] = 1.0f / fmaxf(sqrtf(s), 1e-12f);
}

// ------------------------------------------- centroid bf16 split + |c|^2
__global__ __launch_bounds__(256) void k_prepC(const float* __restrict__ C,
                                               unsigned short* __restrict__ Chi,
                                               unsigned short* __restrict__ Clo,
                                               float* __restrict__ cnorm2,
                                               float* __restrict__ counts) {
  __shared__ float sm[8];
  const int k = blockIdx.x, tid = threadIdx.x;
  float s = 0.f;
#pragma unroll
  for (int j = 0; j < 2; ++j) {
    const int c = tid + j * 256;
    const float v = C[k * D + c];
    s += v * v;
    const unsigned short hi = f2bf(v);
    Chi[k * D + c] = hi;
    Clo[k * D + c] = f2bf(v - bf2f(hi));
  }
#pragma unroll
  for (int m = 16; m; m >>= 1) s += __shfl_xor(s, m, 32);
  if ((tid & 31) == 0) sm[tid >> 5] = s;
  __syncthreads();
  if (tid == 0) {
    float t = 0.f;
#pragma unroll
    for (int i = 0; i < 8; ++i) t += sm[i];
    cnorm2[k] = t;
    counts[k] = 0.f;
  }
}

// --------------------------------- fused distance GEMM (bf16 split-3) + argmin
// Block = 8 waves. Waves split 32 rows (2 strips) x N (4-way). Each wave
// register-blocks 2 N-tiles per iteration, with dual accumulators per tile
// (hi*hi and hi*lo+lo*hi) -> 4 independent WMMA chains per K-step.
__global__ __launch_bounds__(256) void k_dist(const float* __restrict__ X,
                                              const float* __restrict__ invn,
                                              const unsigned short* __restrict__ Chi,
                                              const unsigned short* __restrict__ Clo,
                                              const float* __restrict__ cnorm2,
                                              int* __restrict__ assign,
                                              float* __restrict__ outIdx) {
  __shared__ __align__(16) unsigned short lds[2 * 32 * D];  // Ahi | Alo, 64 KB
  unsigned short* Ahi = lds;
  unsigned short* Alo = lds + 32 * D;
  const int tid = threadIdx.x;
  const int rowBase = blockIdx.x * 32;

  // Stage 32 normalized rows as bf16 hi/lo into LDS.
  for (int g = tid; g < 32 * (D / 4); g += 256) {
    const int r = g >> 7;               // 128 float4 groups per row
    const int c = (g & 127) << 2;
    const float s = invn[rowBase + r];
    float4 x = *(const float4*)(X + (rowBase + r) * D + c);
    x.x *= s; x.y *= s; x.z *= s; x.w *= s;
    const unsigned short h0 = f2bf(x.x), h1 = f2bf(x.y), h2 = f2bf(x.z), h3 = f2bf(x.w);
    const unsigned short l0 = f2bf(x.x - bf2f(h0)), l1 = f2bf(x.y - bf2f(h1));
    const unsigned short l2 = f2bf(x.z - bf2f(h2)), l3 = f2bf(x.w - bf2f(h3));
    *(uint2*)(Ahi + r * D + c) =
        make_uint2((unsigned)h0 | ((unsigned)h1 << 16), (unsigned)h2 | ((unsigned)h3 << 16));
    *(uint2*)(Alo + r * D + c) =
        make_uint2((unsigned)l0 | ((unsigned)l1 << 16), (unsigned)l2 | ((unsigned)l3 << 16));
  }
  __syncthreads();

  const int lane  = tid & 31;
  const int w     = tid >> 5;
  const int wm    = w & 1;      // which 16-row strip of the 32
  const int wn    = w >> 1;     // 4-way split of the N dimension
  const int col16 = lane & 15;
  const int hi16  = lane >> 4;

  // A fragment source (ISA 16-bit A 16x32 layout): groups of 8 K at k0+{0|8} and k0+16+{0|8}
  const unsigned short* aHiP = Ahi + (wm * 16 + col16) * D + hi16 * 8;
  const unsigned short* aLoP = Alo + (wm * 16 + col16) * D + hi16 * 8;

  float best[8];
  int   bidx[8];
#pragma unroll
  for (int r = 0; r < 8; ++r) { best[r] = 3.4e38f; bidx[r] = 0x7fffffff; }

  for (int t = wn; t < K_CL / 16; t += 8) {
    const int n0 = t * 16 + col16;            // this lane's column, tile 0
    const int n1 = (t + 4) * 16 + col16;      // tile 1 (always > n0)
    // B fragment (32x16, 16-bit): groups of 8 K at k0+{0|8} and k0+16+{0|8}, matching A
    const unsigned short* bHiP0 = Chi + n0 * D + hi16 * 8;
    const unsigned short* bLoP0 = Clo + n0 * D + hi16 * 8;
    const unsigned short* bHiP1 = Chi + n1 * D + hi16 * 8;
    const unsigned short* bLoP1 = Clo + n1 * D + hi16 * 8;
    v8f aHH0 = {0.f,0.f,0.f,0.f,0.f,0.f,0.f,0.f};
    v8f aML0 = {0.f,0.f,0.f,0.f,0.f,0.f,0.f,0.f};
    v8f aHH1 = {0.f,0.f,0.f,0.f,0.f,0.f,0.f,0.f};
    v8f aML1 = {0.f,0.f,0.f,0.f,0.f,0.f,0.f,0.f};
#pragma unroll 2
    for (int k0 = 0; k0 < D; k0 += 32) {
      Frag ah, al, bh0, bl0, bh1, bl1;
      ah.q[0]  = *(const uint4*)(aHiP + k0);
      ah.q[1]  = *(const uint4*)(aHiP + k0 + 16);
      al.q[0]  = *(const uint4*)(aLoP + k0);
      al.q[1]  = *(const uint4*)(aLoP + k0 + 16);
      bh0.q[0] = *(const uint4*)(bHiP0 + k0);
      bh0.q[1] = *(const uint4*)(bHiP0 + k0 + 16);
      bl0.q[0] = *(const uint4*)(bLoP0 + k0);
      bl0.q[1] = *(const uint4*)(bLoP0 + k0 + 16);
      bh1.q[0] = *(const uint4*)(bHiP1 + k0);
      bh1.q[1] = *(const uint4*)(bHiP1 + k0 + 16);
      bl1.q[0] = *(const uint4*)(bLoP1 + k0);
      bl1.q[1] = *(const uint4*)(bLoP1 + k0 + 16);
      // 4 independent accumulator chains per step
      aHH0 = __builtin_amdgcn_wmma_f32_16x16x32_bf16(false, ah.v, false, bh0.v, (short)0, aHH0, false, false);
      aHH1 = __builtin_amdgcn_wmma_f32_16x16x32_bf16(false, ah.v, false, bh1.v, (short)0, aHH1, false, false);
      aML0 = __builtin_amdgcn_wmma_f32_16x16x32_bf16(false, ah.v, false, bl0.v, (short)0, aML0, false, false);
      aML1 = __builtin_amdgcn_wmma_f32_16x16x32_bf16(false, ah.v, false, bl1.v, (short)0, aML1, false, false);
      aML0 = __builtin_amdgcn_wmma_f32_16x16x32_bf16(false, al.v, false, bh0.v, (short)0, aML0, false, false);
      aML1 = __builtin_amdgcn_wmma_f32_16x16x32_bf16(false, al.v, false, bh1.v, (short)0, aML1, false, false);
    }
    const float cn0 = cnorm2[n0];
    const float cn1 = cnorm2[n1];
#pragma unroll
    for (int r = 0; r < 8; ++r) {       // n0 first, then n1: keeps first-min order
      const float d0 = cn0 - 2.0f * (aHH0[r] + aML0[r]);
      if (d0 < best[r]) { best[r] = d0; bidx[r] = n0; }
      const float d1 = cn1 - 2.0f * (aHH1[r] + aML1[r]);
      if (d1 < best[r]) { best[r] = d1; bidx[r] = n1; }
    }
  }

  // Reduce across the 16 lanes of each half-wave (keep lowest index on ties).
#pragma unroll
  for (int m = 1; m <= 8; m <<= 1) {
#pragma unroll
    for (int r = 0; r < 8; ++r) {
      const float od = __shfl_xor(best[r], m, 32);
      const int   oi = __shfl_xor(bidx[r], m, 32);
      if (od < best[r] || (od == best[r] && oi < bidx[r])) { best[r] = od; bidx[r] = oi; }
    }
  }

  __syncthreads();                     // A tile no longer needed; reuse LDS
  float* redD = (float*)lds;           // [8 waves][16 rows-in-strip]
  int*   redI = (int*)lds + 128;
  if (col16 == 0) {                    // lanes 0 and 16 hold the half-wave results
#pragma unroll
    for (int r = 0; r < 8; ++r) {
      redD[w * 16 + hi16 * 8 + r] = best[r];
      redI[w * 16 + hi16 * 8 + r] = bidx[r];
    }
  }
  __syncthreads();
  if (tid < 32) {
    const int lr = tid;                // local row 0..31
    const int wmr = lr >> 4, rr = lr & 15;
    float bd = 3.4e38f;
    int   bi = 0x7fffffff;
#pragma unroll
    for (int j = 0; j < 4; ++j) {      // combine the 4 N-split waves of this strip
      const int ww = wmr + 2 * j;
      const float d = redD[ww * 16 + rr];
      const int   i = redI[ww * 16 + rr];
      if (d < bd || (d == bd && i < bi)) { bd = d; bi = i; }
    }
    assign[rowBase + lr] = bi;
    outIdx[rowBase + lr] = (float)bi;
  }
}

// ---------------------------------------------------------------- histogram
__global__ __launch_bounds__(256) void k_hist(const int* __restrict__ assign,
                                              float* __restrict__ counts) {
  const int i = blockIdx.x * 256 + threadIdx.x;
  atomicAdd(&counts[assign[i]], 1.0f);
}

// --------------------------------------------- EMA size + Laplace renorm
__global__ __launch_bounds__(1024) void k_newsize(const float* __restrict__ es,
                                                  const float* __restrict__ counts,
                                                  float* __restrict__ newS) {
  __shared__ float sm[1024];
  const int tid = threadIdx.x;
  const float OMD = (float)(1.0 - 0.99);
  const float va = es[tid] * DECAYF + OMD * counts[tid];
  const float vb = es[tid + 1024] * DECAYF + OMD * counts[tid + 1024];
  sm[tid] = va + vb;
  __syncthreads();
  for (int s = 512; s > 0; s >>= 1) {
    if (tid < s) sm[tid] += sm[tid + s];
    __syncthreads();
  }
  const float n = sm[0];
  const float denom = n + (float)K_CL * 1e-5f;
  newS[tid]        = (va + 1e-5f) / denom * n;
  newS[tid + 1024] = (vb + 1e-5f) / denom * n;
}

// ---------------------------------------------------------------- new_w init
__global__ __launch_bounds__(256) void k_initw(const float4* __restrict__ emaW,
                                               float4* __restrict__ newW) {
  const int i = blockIdx.x * 256 + threadIdx.x;
  float4 v = emaW[i];
  v.x *= DECAYF; v.y *= DECAYF; v.z *= DECAYF; v.w *= DECAYF;
  newW[i] = v;
}

// ------------------------------------------------------- scatter dw into new_w
__global__ __launch_bounds__(128) void k_scatter(const float* __restrict__ X,
                                                 const int* __restrict__ assign,
                                                 float* __restrict__ newW) {
  const float OMD = (float)(1.0 - 0.99);
  const int i = blockIdx.x;
  const int a = assign[i];
  const float4 v = ((const float4*)(X + i * D))[threadIdx.x];
  float* wv = newW + a * D + threadIdx.x * 4;
  atomicAdd(&wv[0], OMD * v.x);
  atomicAdd(&wv[1], OMD * v.y);
  atomicAdd(&wv[2], OMD * v.z);
  atomicAdd(&wv[3], OMD * v.w);
}

// -------------------------------------------------------- new_centroids
__global__ __launch_bounds__(256) void k_finalC(const float4* __restrict__ newW,
                                                const float* __restrict__ newS,
                                                float4* __restrict__ newC) {
  const int i = blockIdx.x * 256 + threadIdx.x;  // float4 index
  const int k = i >> 7;                          // 128 float4 per row
  const float s = newS[k];
  float4 v = newW[i];
  v.x /= s; v.y /= s; v.z /= s; v.w /= s;
  newC[i] = v;
}

// ---------------------------------------------------------------- quantized
__global__ __launch_bounds__(128) void k_quant(const float* __restrict__ C,
                                               const int* __restrict__ assign,
                                               float* __restrict__ Q) {
  const int i = blockIdx.x;
  const int a = assign[i];
  ((float4*)(Q + i * D))[threadIdx.x] = ((const float4*)(C + a * D))[threadIdx.x];
}

extern "C" void kernel_launch(void* const* d_in, const int* in_sizes, int n_in,
                              void* d_out, int out_size, void* d_ws, size_t ws_size,
                              hipStream_t stream) {
  (void)in_sizes; (void)n_in; (void)out_size; (void)ws_size;
  const float* X        = (const float*)d_in[0];
  const float* C        = (const float*)d_in[1];
  const float* emaSize  = (const float*)d_in[2];
  const float* emaW     = (const float*)d_in[3];

  float* out    = (float*)d_out;
  float* quant  = out;                       // [N, D]
  float* outIdx = quant + (size_t)N_PTS * D; // [N, 1]
  float* newC   = outIdx + N_PTS;            // [K, D]
  float* newS   = newC + (size_t)K_CL * D;   // [K]
  float* newW   = newS + K_CL;               // [K, D]

  float* invn            = (float*)d_ws;                 // 32768
  float* cnorm2          = invn + N_PTS;                 // 2048
  float* counts          = cnorm2 + K_CL;                // 2048
  int*   assign          = (int*)(counts + K_CL);        // 32768
  unsigned short* Chi    = (unsigned short*)(assign + N_PTS);
  unsigned short* Clo    = Chi + (size_t)K_CL * D;

  k_norms  <<<N_PTS / 8, 256, 0, stream>>>(X, invn);
  k_prepC  <<<K_CL, 256, 0, stream>>>(C, Chi, Clo, cnorm2, counts);
  k_dist   <<<N_PTS / 32, 256, 0, stream>>>(X, invn, Chi, Clo, cnorm2, assign, outIdx);
  k_hist   <<<N_PTS / 256, 256, 0, stream>>>(assign, counts);
  k_newsize<<<1, 1024, 0, stream>>>(emaSize, counts, newS);
  k_initw  <<<(K_CL * D / 4) / 256, 256, 0, stream>>>((const float4*)emaW, (float4*)newW);
  k_scatter<<<N_PTS, 128, 0, stream>>>(X, assign, newW);
  k_finalC <<<(K_CL * D / 4) / 256, 256, 0, stream>>>((const float4*)newW, newS, (float4*)newC);
  k_quant  <<<N_PTS, 128, 0, stream>>>(C, assign, quant);
}
